// OTLoss_6485400617245
// MI455X (gfx1250) — compile-verified, matrix operand order checked by reading, not run on hardware
//
#include <hip/hip_runtime.h>
#include <hip/hip_bf16.h>
#include <math.h>
#include <stdint.h>

#define B_N 4096
#define D_K 1024
#define EPS_REG 0.05f
#define NUM_ITER 100

typedef __attribute__((ext_vector_type(16))) _Float16 v16h;
typedef __attribute__((ext_vector_type(8)))  _Float16 v8h;
typedef __attribute__((ext_vector_type(4)))  _Float16 v4h;
typedef __attribute__((ext_vector_type(8)))  float    v8f;
typedef __attribute__((ext_vector_type(4)))  float    v4f;

// ---------- CDNA5 async global->LDS copy (ASYNCcnt path) ----------
__device__ __forceinline__ void async_copy_b128(uint32_t lds_addr, const _Float16* gptr) {
    asm volatile("global_load_async_to_lds_b128 %0, %1, off"
                 :: "v"(lds_addr), "v"(gptr)
                 : "memory");
}
__device__ __forceinline__ void wait_async0() {
    asm volatile("s_wait_asynccnt 0x0" ::: "memory");
}

// ---------------- init u, v, max ----------------
__global__ void k_init(float* u, float* v, int* maxBits) {
    int i = blockIdx.x * blockDim.x + threadIdx.x;
    if (i < B_N) { u[i] = 1.0f / B_N; v[i] = 1.0f / B_N; }
    if (i == 0) *maxBits = 0;
}

// ---------------- row-normalize fp32 -> unit-norm f16 (one b128 load per thread) ----
__global__ __launch_bounds__(256) void k_normalize(const float* __restrict__ audio,
                                                   const float* __restrict__ text,
                                                   _Float16* __restrict__ Xn,
                                                   _Float16* __restrict__ Yn) {
    __shared__ float red[256];
    int r = blockIdx.x;
    int row = r & (B_N - 1);
    const float* src = (r < B_N ? audio : text) + (size_t)row * D_K;
    _Float16*    dst = (r < B_N ? Xn    : Yn ) + (size_t)row * D_K;
    v4f x = ((const v4f*)src)[threadIdx.x];           // 256 threads * 4 = 1024
    float ss = x[0]*x[0] + x[1]*x[1] + x[2]*x[2] + x[3]*x[3];
    red[threadIdx.x] = ss; __syncthreads();
    for (int s = 128; s > 0; s >>= 1) {
        if (threadIdx.x < s) red[threadIdx.x] += red[threadIdx.x + s];
        __syncthreads();
    }
    float inv = rsqrtf(red[0] + 1e-30f);
    v4h h;
    h[0] = (_Float16)(x[0] * inv); h[1] = (_Float16)(x[1] * inv);
    h[2] = (_Float16)(x[2] * inv); h[3] = (_Float16)(x[3] * inv);
    ((v4h*)dst)[threadIdx.x] = h;
}

// ---------------- WMMA GEMM: M = 1 - Xn @ Yn^T ; track max(M) ----------------
// Block = 128 threads (4 waves), block tile 64x64, wave tile 16x64.
// B tile (64 cols x 32 k, f16) staged in LDS via async copy, double buffered.
__device__ __forceinline__ v16h load_a(const _Float16* aRow, int kb) {
    v8h lo = *(const v8h*)(aRow + kb);
    v8h hi = *(const v8h*)(aRow + kb + 16);
    return __builtin_shufflevector(lo, hi, 0,1,2,3,4,5,6,7,8,9,10,11,12,13,14,15);
}
__device__ __forceinline__ v16h combine16(v8h lo, v8h hi) {
    return __builtin_shufflevector(lo, hi, 0,1,2,3,4,5,6,7,8,9,10,11,12,13,14,15);
}
// thread t copies chunks 2t, 2t+1 of the 64x32 tile (chunk = 16B; col = c>>2, qk = c&3)
__device__ __forceinline__ void stage_btile(const _Float16* __restrict__ Yn,
                                            int colBase, int k, uint32_t ldsBase, int t) {
    int c0 = 2 * t;
    const _Float16* g0 = Yn + (size_t)(colBase + (c0 >> 2)) * D_K + k + (c0 & 3) * 8;
    uint32_t l0 = ldsBase + (uint32_t)c0 * 16u;
    async_copy_b128(l0,       g0);
    async_copy_b128(l0 + 16u, g0 + 8);
}

__global__ __launch_bounds__(128) void k_gemm_cossim(const _Float16* __restrict__ Xn,
                                                     const _Float16* __restrict__ Yn,
                                                     float* __restrict__ Mmat,
                                                     int* __restrict__ maxBits) {
    __shared__ __align__(16) _Float16 btile[2][64 * 32];
    __shared__ float red[128];
    const int t    = threadIdx.x;
    const int lane = t & 31;
    const int wave = t >> 5;
    const int l15  = lane & 15;
    const int hi   = lane >> 4;
    const int rowBase = blockIdx.y * 64 + wave * 16;
    const int colBase = blockIdx.x * 64;
    const int khalf = hi * 8;                  // ISA 16-bit 16x32 operand layout

    const uint32_t ldsB0 = (uint32_t)(uintptr_t)&btile[0][0];
    const uint32_t ldsB1 = (uint32_t)(uintptr_t)&btile[1][0];
    const _Float16* aRow = Xn + (size_t)(rowBase + l15) * D_K;

    stage_btile(Yn, colBase, 0, ldsB0, t);     // prime tile 0
    v16h aCur = load_a(aRow, khalf);           // prime A regs (k=0)

    v8f acc0 = {}, acc1 = {}, acc2 = {}, acc3 = {};

    for (int step = 0; step < D_K / 32; ++step) {
        wait_async0();                         // my portion of current tile landed
        __syncthreads();                       // whole tile landed
        const int buf = step & 1;
        v16h aNext = aCur;
        if (step != D_K / 32 - 1) {
            stage_btile(Yn, colBase, (step + 1) * 32, buf ? ldsB0 : ldsB1, t);
            aNext = load_a(aRow, (step + 1) * 32 + khalf);
        }
        const v8h* bt = (const v8h*)&btile[buf][0];   // per-col stride = 4 v8h
        int ci;
        ci = (0 * 16 + l15) * 4; v16h b0 = combine16(bt[ci + hi], bt[ci + hi + 2]);
        ci = (1 * 16 + l15) * 4; v16h b1 = combine16(bt[ci + hi], bt[ci + hi + 2]);
        ci = (2 * 16 + l15) * 4; v16h b2 = combine16(bt[ci + hi], bt[ci + hi + 2]);
        ci = (3 * 16 + l15) * 4; v16h b3 = combine16(bt[ci + hi], bt[ci + hi + 2]);

        acc0 = __builtin_amdgcn_wmma_f32_16x16x32_f16(false, aCur, false, b0, (short)0, acc0, false, false);
        acc1 = __builtin_amdgcn_wmma_f32_16x16x32_f16(false, aCur, false, b1, (short)0, acc1, false, false);
        acc2 = __builtin_amdgcn_wmma_f32_16x16x32_f16(false, aCur, false, b2, (short)0, acc2, false, false);
        acc3 = __builtin_amdgcn_wmma_f32_16x16x32_f16(false, aCur, false, b3, (short)0, acc3, false, false);
        aCur = aNext;
    }

    float lmax = 0.f;
    const int outRow0 = rowBase + hi * 8;      // C/D layout: vgpr r -> row r (+8 hi half)
#define OT_EMIT(ACC, CI)                                                 \
    do {                                                                 \
        int col = colBase + (CI) * 16 + l15;                             \
        _Pragma("unroll")                                                \
        for (int r = 0; r < 8; ++r) {                                    \
            float mval = 1.0f - (ACC)[r];                                \
            Mmat[(size_t)(outRow0 + r) * B_N + col] = mval;              \
            lmax = fmaxf(lmax, mval);                                    \
        }                                                                \
    } while (0)
    OT_EMIT(acc0, 0); OT_EMIT(acc1, 1); OT_EMIT(acc2, 2); OT_EMIT(acc3, 3);
#undef OT_EMIT

    red[t] = lmax; __syncthreads();
    for (int s = 64; s > 0; s >>= 1) {
        if (t < s) red[t] = fmaxf(red[t], red[t + s]);
        __syncthreads();
    }
    if (t == 0) atomicMax(maxBits, __float_as_int(red[0])); // M >= 0: int order == float order
}

// ---------------- K = exp(-M / (eps * maxM)), in place, b128 ----------------
__global__ __launch_bounds__(256) void k_expK(float* __restrict__ K, const int* __restrict__ maxBits) {
    float scale = -1.0f / (EPS_REG * __int_as_float(*maxBits));
    v4f* K4 = (v4f*)K;
    size_t n4 = (size_t)B_N * B_N / 4;
    for (size_t i = (size_t)blockIdx.x * blockDim.x + threadIdx.x; i < n4;
         i += (size_t)gridDim.x * blockDim.x) {
        v4f x = K4[i];
        v4f r;
        r[0] = expf(x[0] * scale); r[1] = expf(x[1] * scale);
        r[2] = expf(x[2] * scale); r[3] = expf(x[3] * scale);
        K4[i] = r;
    }
}

// ---------------- u = a / (K v)  (block per row, b128 loads) ----------------
__global__ __launch_bounds__(256) void k_rowMatvec(const float* __restrict__ K,
                                                   const float* __restrict__ v,
                                                   float* __restrict__ u) {
    __shared__ float red[256];
    int i = blockIdx.x;
    const v4f* Krow = (const v4f*)(K + (size_t)i * B_N);
    const v4f* v4p  = (const v4f*)v;
    v4f s4 = {};
    for (int c = threadIdx.x; c < B_N / 4; c += 256) s4 += Krow[c] * v4p[c];
    red[threadIdx.x] = s4[0] + s4[1] + s4[2] + s4[3];
    __syncthreads();
    for (int s = 128; s > 0; s >>= 1) {
        if (threadIdx.x < s) red[threadIdx.x] += red[threadIdx.x + s];
        __syncthreads();
    }
    if (threadIdx.x == 0) u[i] = (1.0f / B_N) / red[0];
}

// ------- partial K^T u: 32 row-chunks x 4 col-groups, b128 column loads -------
__global__ __launch_bounds__(256) void k_colPartial(const float* __restrict__ K,
                                                    const float* __restrict__ u,
                                                    float* __restrict__ partial) {
    __shared__ float su[128];
    int g     = blockIdx.x & 3;            // 1024-column group
    int chunk = blockIdx.x >> 2;           // 0..31 -> 128 rows each
    int row0  = chunk * 128;
    if (threadIdx.x < 128) su[threadIdx.x] = u[row0 + threadIdx.x];
    __syncthreads();
    const v4f* Kv = (const v4f*)K;         // row stride = 1024 v4f
    size_t colv = (size_t)g * 256 + threadIdx.x;
    v4f acc = {};
    for (int r = 0; r < 128; ++r)
        acc += Kv[(size_t)(row0 + r) * (B_N / 4) + colv] * su[r];
    ((v4f*)partial)[(size_t)chunk * (B_N / 4) + colv] = acc;
}

// ---------------- v = b / sum(32 partials) ----------------
__global__ void k_colReduce(const float* __restrict__ partial, float* __restrict__ v) {
    int j = blockIdx.x * blockDim.x + threadIdx.x;
    float s = 0.f;
    for (int c = 0; c < 32; ++c) s += partial[(size_t)c * B_N + j];
    v[j] = (1.0f / B_N) / s;
}

// ---------------- per-row loss: -(log_softmax(pi)[i,i]) ----------------
__global__ __launch_bounds__(256) void k_lossRow(const float* __restrict__ K,
                                                 const float* __restrict__ u,
                                                 const float* __restrict__ v,
                                                 float* __restrict__ rowloss) {
    __shared__ __align__(16) float rowbuf[B_N];
    __shared__ float red[256];
    int i = blockIdx.x;
    float ui = u[i];
    const v4f* Krow = (const v4f*)(K + (size_t)i * B_N);
    const v4f* v4p  = (const v4f*)v;
    float lmax = -1e30f;
    for (int c = threadIdx.x; c < B_N / 4; c += 256) {
        v4f tt = (Krow[c] * v4p[c]) * ui;
        ((v4f*)rowbuf)[c] = tt;
        lmax = fmaxf(lmax, fmaxf(fmaxf(tt[0], tt[1]), fmaxf(tt[2], tt[3])));
    }
    red[threadIdx.x] = lmax; __syncthreads();
    for (int s = 128; s > 0; s >>= 1) {
        if (threadIdx.x < s) red[threadIdx.x] = fmaxf(red[threadIdx.x], red[threadIdx.x + s]);
        __syncthreads();
    }
    float m = red[0];
    __syncthreads();
    float se = 0.f;
    for (int j = threadIdx.x; j < B_N; j += 256) se += expf(rowbuf[j] - m);
    red[threadIdx.x] = se; __syncthreads();
    for (int s = 128; s > 0; s >>= 1) {
        if (threadIdx.x < s) red[threadIdx.x] += red[threadIdx.x + s];
        __syncthreads();
    }
    if (threadIdx.x == 0) {
        float logZ = m + logf(red[0]);
        rowloss[i] = -(rowbuf[i] - logZ);
    }
}

// ---------------- deterministic final mean ----------------
__global__ __launch_bounds__(256) void k_finalLoss(const float* __restrict__ rowloss,
                                                   float* __restrict__ out) {
    __shared__ float red[256];
    float s = 0.f;
    for (int i = threadIdx.x; i < B_N; i += 256) s += rowloss[i];
    red[threadIdx.x] = s; __syncthreads();
    for (int t = 128; t > 0; t >>= 1) {
        if (threadIdx.x < t) red[threadIdx.x] += red[threadIdx.x + t];
        __syncthreads();
    }
    if (threadIdx.x == 0) out[0] = red[0] / B_N;
}

extern "C" void kernel_launch(void* const* d_in, const int* in_sizes, int n_in,
                              void* d_out, int out_size, void* d_ws, size_t ws_size,
                              hipStream_t stream) {
    (void)in_sizes; (void)n_in; (void)out_size; (void)ws_size;
    const float* audio = (const float*)d_in[0];
    const float* text  = (const float*)d_in[1];

    char* ws = (char*)d_ws;
    size_t off = 0;
    _Float16* Xn  = (_Float16*)(ws + off); off += (size_t)B_N * D_K * sizeof(_Float16);
    _Float16* Yn  = (_Float16*)(ws + off); off += (size_t)B_N * D_K * sizeof(_Float16);
    float* Kmat   = (float*)(ws + off);    off += (size_t)B_N * B_N * sizeof(float);
    int*   maxB   = (int*)(ws + off);      off += 256;
    float* u      = (float*)(ws + off);    off += (size_t)B_N * sizeof(float);
    float* v      = (float*)(ws + off);    off += (size_t)B_N * sizeof(float);
    float* part   = (float*)(ws + off);    off += (size_t)32 * B_N * sizeof(float);
    float* rloss  = (float*)(ws + off);    off += (size_t)B_N * sizeof(float);

    k_init<<<B_N / 256, 256, 0, stream>>>(u, v, maxB);
    k_normalize<<<2 * B_N, 256, 0, stream>>>(audio, text, Xn, Yn);

    dim3 gGemm(B_N / 64, B_N / 64);
    k_gemm_cossim<<<gGemm, 128, 0, stream>>>(Xn, Yn, Kmat, maxB);
    k_expK<<<8192, 256, 0, stream>>>(Kmat, maxB);

    for (int it = 0; it < NUM_ITER; ++it) {
        k_rowMatvec<<<B_N, 256, 0, stream>>>(Kmat, v, u);       // u = a/(K v)
        k_colPartial<<<128, 256, 0, stream>>>(Kmat, u, part);   // K^T u partials
        k_colReduce<<<B_N / 256, 256, 0, stream>>>(part, v);    // v = b/(K^T u)
    }

    k_lossRow<<<B_N, 256, 0, stream>>>(Kmat, u, v, rloss);
    k_finalLoss<<<1, 256, 0, stream>>>(rloss, (float*)d_out);
}
